// BaselineRNN_23270132810441
// MI455X (gfx1250) — matrix operand order repaired and checked by reading, not compiled, not run on hardware
//
#include <hip/hip_runtime.h>
#include <stdint.h>

#define Bsz 256
#define Tsz 512
#define Dsz 1024
#define Hsz 1024

typedef __attribute__((ext_vector_type(16))) __bf16 v16bf;
typedef __attribute__((ext_vector_type(8)))  float  v8f;

__device__ __forceinline__ unsigned short f2bf(float f) {
  union { float f; uint32_t u; } x; x.f = f;
  uint32_t r = x.u + 0x7FFFu + ((x.u >> 16) & 1u);   // round-to-nearest-even
  return (unsigned short)(r >> 16);
}

__device__ __forceinline__ uint32_t pack2bf(float lo, float hi) {
  return (uint32_t)f2bf(lo) | ((uint32_t)f2bf(hi) << 16);
}

// gfx1250 async DMA: global -> LDS, 16 bytes per lane, tracked by ASYNCcnt.
__device__ __forceinline__ void async_ld_b128(uint32_t lds_addr, const void* gptr) {
  asm volatile("global_load_async_to_lds_b128 %0, %1, off"
               :: "v"(lds_addr), "v"(gptr)
               : "memory");
}
__device__ __forceinline__ void wait_async0() {
  asm volatile("s_wait_asynccnt 0x0" ::: "memory");
}

// ---------------- one-shot prep kernels ----------------

__global__ void convert_weights(const float* __restrict__ Wi2h,
                                const float* __restrict__ Wh2o,
                                unsigned short* __restrict__ Wxb,
                                unsigned short* __restrict__ Whb,
                                unsigned short* __restrict__ Whob) {
  int idx = blockIdx.x * blockDim.x + threadIdx.x;   // 0 .. 1024*1024-1
  int j = idx >> 10;
  int d = idx & 1023;
  Wxb[idx]  = f2bf(Wi2h[(size_t)j * 2048 + d]);
  Whb[idx]  = f2bf(Wi2h[(size_t)j * 2048 + 1024 + d]);
  Whob[idx] = f2bf(Wh2o[idx]);
}

__global__ void zero_f32(float* __restrict__ p) {
  int idx = blockIdx.x * blockDim.x + threadIdx.x;
  p[idx] = 0.0f;
}

__global__ void copy_f32(const float* __restrict__ src, float* __restrict__ dst) {
  int idx = blockIdx.x * blockDim.x + threadIdx.x;
  dst[idx] = src[idx];
}

// ---------------- per-timestep recurrence GEMM ----------------
// C(256x1024) = tanh( X_t(256x1024)*Wx^T + H(256x1024)*Wh^T + b )
// Grid (16,4): 64x64 output tile per WG; 8 waves, each owns a 16x32 slice.
// A tile: fp32->bf16 VALU staging (double-buffered LDS).
// W tile: global_load_async_to_lds_b128 DMA (double-buffered LDS).

__launch_bounds__(256, 1)
__global__ void rnn_step(const float* __restrict__ seq, int t,
                         const float* __restrict__ hprev,
                         float* __restrict__ hnext,
                         const unsigned short* __restrict__ Wxb,
                         const unsigned short* __restrict__ Whb,
                         const float* __restrict__ bias) {
  __shared__ unsigned short As[2][64][68];   // rows x K, padded (4B-aligned dword IO)
  __shared__ unsigned short Ws[2][64][72];   // cols x K, 144B rows (16B-aligned for B128 DMA)

  const int tid  = threadIdx.x;
  const int lane = tid & 31;
  const int wv   = tid >> 5;
  const int row0 = blockIdx.y * 64;
  const int col0 = blockIdx.x * 64;
  const int mrow = (wv & 3) * 16;
  const int ncol = (wv >> 2) * 32;

  v8f c0 = {};
  v8f c1 = {};

  const int srow = tid >> 2;          // staging: 4 threads per row
  const int skk  = (tid & 3) * 16;    // each covers 16 consecutive K

  auto stageA = [&](int buf, int kc) {
    const bool fromx = (kc < Dsz);
    const float* asrc = fromx
        ? (seq   + (size_t)(row0 + srow) * ((size_t)Tsz * Dsz) + (size_t)t * Dsz + kc)
        : (hprev + (size_t)(row0 + srow) * Hsz + (kc - Dsz));
    const float4* a4 = (const float4*)(asrc + skk);
#pragma unroll
    for (int i = 0; i < 4; ++i) {
      float4 f = a4[i];
      *(uint32_t*)&As[buf][srow][skk + 4 * i]     = pack2bf(f.x, f.y);
      *(uint32_t*)&As[buf][srow][skk + 4 * i + 2] = pack2bf(f.z, f.w);
    }
  };

  auto stageW = [&](int buf, int kc) {
    const unsigned short* wbase =
        ((kc < Dsz) ? Wxb : Whb) + (size_t)col0 * 1024 + (kc & 1023);
#pragma unroll
    for (int r = 0; r < 2; ++r) {
      int ti  = tid + 256 * r;         // 512 x 16B transfers cover the 8KB tile
      int row = ti >> 3;
      int seg = ti & 7;
      async_ld_b128((uint32_t)(uintptr_t)&Ws[buf][row][seg * 8],
                    wbase + (size_t)row * 1024 + seg * 8);
    }
  };

  stageA(0, 0);
  stageW(0, 0);
  wait_async0();
  __syncthreads();

  for (int kc = 0; kc < Dsz + Hsz; kc += 64) {
    const int cur = (kc >> 6) & 1;
    const int nxt = cur ^ 1;
    if (kc + 64 < Dsz + Hsz) {         // prefetch next chunk while computing
      stageA(nxt, kc + 64);
      stageW(nxt, kc + 64);
    }

#pragma unroll
    for (int kh = 0; kh < 64; kh += 32) {
      union { v16bf vec; uint32_t u[8]; } a, b0, b1;
      const int khi = (lane >= 16) ? 16 : 0;
      const int m   = lane & 15;
#pragma unroll
      for (int v = 0; v < 8; ++v) {
        // A 16x32 layout (contiguous K, matches B): VGPR v holds K=2v,2v+1 (+16 hi-half)
        int kb = 2 * v + khi;
        a.u[v] = *(const uint32_t*)&As[cur][mrow + m][kh + kb];
      }
      const int kbb = (lane >= 16) ? 16 : 0;
#pragma unroll
      for (int v = 0; v < 8; ++v) {
        // B 32x16 layout: lane = N, VGPR v holds K=2v,2v+1 (lo) / 16+2v,17+2v (hi)
        b0.u[v] = *(const uint32_t*)&Ws[cur][ncol + m][kh + kbb + 2 * v];
        b1.u[v] = *(const uint32_t*)&Ws[cur][ncol + 16 + m][kh + kbb + 2 * v];
      }
      c0 = __builtin_amdgcn_wmma_f32_16x16x32_bf16(false, a.vec, false, b0.vec,
                                                   (short)0, c0, false, false);
      c1 = __builtin_amdgcn_wmma_f32_16x16x32_bf16(false, a.vec, false, b1.vec,
                                                   (short)0, c1, false, false);
    }

    wait_async0();      // DMA for next chunk has the whole WMMA block to land
    __syncthreads();
  }

  // Epilogue: bias + tanh, write fp32 hidden. C layout: hi-half lanes -> M+=8, N=lane&15.
  const int m0  = (lane >= 16) ? 8 : 0;
  const int n   = lane & 15;
  const int gc0 = col0 + ncol + n;
  const int gc1 = gc0 + 16;
  const float bi0 = bias[gc0];
  const float bi1 = bias[gc1];
#pragma unroll
  for (int v = 0; v < 8; ++v) {
    int gr = row0 + mrow + m0 + v;
    hnext[(size_t)gr * Hsz + gc0] = tanhf(c0[v] + bi0);
    hnext[(size_t)gr * Hsz + gc1] = tanhf(c1[v] + bi1);
  }
}

// ---------------- final projection: out = h * W_h2o^T + b (no tanh) ----------------

__launch_bounds__(256, 1)
__global__ void out_gemm(const float* __restrict__ hfin,
                         const unsigned short* __restrict__ Whob,
                         const float* __restrict__ bias,
                         float* __restrict__ out) {
  __shared__ unsigned short As[64][68];
  __shared__ unsigned short Ws[64][72];

  const int tid  = threadIdx.x;
  const int lane = tid & 31;
  const int wv   = tid >> 5;
  const int row0 = blockIdx.y * 64;
  const int col0 = blockIdx.x * 64;
  const int mrow = (wv & 3) * 16;
  const int ncol = (wv >> 2) * 32;

  v8f c0 = {};
  v8f c1 = {};

  const int srow = tid >> 2;
  const int skk  = (tid & 3) * 16;

  for (int kc = 0; kc < Hsz; kc += 64) {
    const float4* a4 = (const float4*)(hfin + (size_t)(row0 + srow) * Hsz + kc + skk);
#pragma unroll
    for (int i = 0; i < 4; ++i) {
      float4 f = a4[i];
      *(uint32_t*)&As[srow][skk + 4 * i]     = pack2bf(f.x, f.y);
      *(uint32_t*)&As[srow][skk + 4 * i + 2] = pack2bf(f.z, f.w);
    }
    const unsigned short* wbase = Whob + (size_t)col0 * 1024 + kc;
#pragma unroll
    for (int r = 0; r < 2; ++r) {
      int ti  = tid + 256 * r;
      int row = ti >> 3;
      int seg = ti & 7;
      async_ld_b128((uint32_t)(uintptr_t)&Ws[row][seg * 8],
                    wbase + (size_t)row * 1024 + seg * 8);
    }
    wait_async0();
    __syncthreads();

#pragma unroll
    for (int kh = 0; kh < 64; kh += 32) {
      union { v16bf vec; uint32_t u[8]; } a, b0, b1;
      const int khi = (lane >= 16) ? 16 : 0;
      const int m   = lane & 15;
#pragma unroll
      for (int v = 0; v < 8; ++v) {
        int kb = 2 * v + khi;
        a.u[v] = *(const uint32_t*)&As[mrow + m][kh + kb];
      }
      const int kbb = (lane >= 16) ? 16 : 0;
#pragma unroll
      for (int v = 0; v < 8; ++v) {
        b0.u[v] = *(const uint32_t*)&Ws[ncol + m][kh + kbb + 2 * v];
        b1.u[v] = *(const uint32_t*)&Ws[ncol + 16 + m][kh + kbb + 2 * v];
      }
      c0 = __builtin_amdgcn_wmma_f32_16x16x32_bf16(false, a.vec, false, b0.vec,
                                                   (short)0, c0, false, false);
      c1 = __builtin_amdgcn_wmma_f32_16x16x32_bf16(false, a.vec, false, b1.vec,
                                                   (short)0, c1, false, false);
    }
    __syncthreads();
  }

  const int m0  = (lane >= 16) ? 8 : 0;
  const int n   = lane & 15;
  const int gc0 = col0 + ncol + n;
  const int gc1 = gc0 + 16;
  const float bi0 = bias[gc0];
  const float bi1 = bias[gc1];
#pragma unroll
  for (int v = 0; v < 8; ++v) {
    int gr = row0 + mrow + m0 + v;
    out[(size_t)gr * Dsz + gc0] = c0[v] + bi0;
    out[(size_t)gr * Dsz + gc1] = c1[v] + bi1;
  }
}

// ---------------- host-side orchestration ----------------

extern "C" void kernel_launch(void* const* d_in, const int* in_sizes, int n_in,
                              void* d_out, int out_size, void* d_ws, size_t ws_size,
                              hipStream_t stream) {
  (void)in_sizes; (void)n_in; (void)out_size; (void)ws_size;

  const float* seq  = (const float*)d_in[0];   // (256, 512, 1024) f32
  const float* Wi2h = (const float*)d_in[1];   // (1024, 2048) f32
  const float* bi2h = (const float*)d_in[2];   // (1024,) f32
  const float* Wh2o = (const float*)d_in[3];   // (1024, 1024) f32
  const float* bh2o = (const float*)d_in[4];   // (1024,) f32
  float* out = (float*)d_out;                  // [256*1024 output | 256*1024 hidden]

  // Workspace (8 MB): bf16 weights + fp32 hidden ping-pong
  unsigned short* Wxb  = (unsigned short*)d_ws;          // 2 MB
  unsigned short* Whb  = Wxb  + (size_t)1024 * 1024;     // 2 MB
  unsigned short* Whob = Whb  + (size_t)1024 * 1024;     // 2 MB
  float* hA = (float*)(Whob + (size_t)1024 * 1024);      // 1 MB
  float* hB = hA + (size_t)Bsz * Hsz;                    // 1 MB

  convert_weights<<<4096, 256, 0, stream>>>(Wi2h, Wh2o, Wxb, Whb, Whob);
  zero_f32<<<(Bsz * Hsz) / 256, 256, 0, stream>>>(hA);

  float* hp = hA;
  float* hn = hB;
  for (int t = 0; t < Tsz; ++t) {
    rnn_step<<<dim3(16, 4), 256, 0, stream>>>(seq, t, hp, hn, Wxb, Whb, bi2h);
    float* tmp = hp; hp = hn; hn = tmp;
  }

  out_gemm<<<dim3(16, 4), 256, 0, stream>>>(hp, Whob, bh2o, out);
  copy_f32<<<(Bsz * Hsz) / 256, 256, 0, stream>>>(hp, out + (size_t)Bsz * Hsz);
}
